// LetterGNN_88682484727907
// MI455X (gfx1250) — matrix operand order repaired and checked by reading, not compiled, not run on hardware
//
#include <hip/hip_runtime.h>
#include <hip/hip_bf16.h>

typedef __attribute__((ext_vector_type(16))) _Float16 v16h;
typedef __attribute__((ext_vector_type(4)))  _Float16 v4h;
typedef __attribute__((ext_vector_type(8)))  float    v8f;

#define NN     100000
#define DHID   128
#define DFF    256
#define NCLS   26
#define NGRAPH 512

union FragH { v16h v; uint4 u[2]; };

// ---------------- utility kernels ----------------

__global__ void k_cvt_f16(const float* __restrict__ in, _Float16* __restrict__ out, int n) {
    int i = blockIdx.x * blockDim.x + threadIdx.x;
    if (i < n) out[i] = (_Float16)in[i];
}

__global__ void k_fill(float* __restrict__ p, float v, int n) {
    int i = blockIdx.x * blockDim.x + threadIdx.x;
    if (i < n) p[i] = v;
}

// degree accumulation over edge destinations (self-loops pre-seeded via k_fill(1))
__global__ void k_deg(const int* __restrict__ dst, float* __restrict__ deg, int E) {
    int i = blockIdx.x * blockDim.x + threadIdx.x;
    if (i < E) atomicAdd(&deg[dst[i]], 1.0f);
}

__global__ void k_rsqrt_inplace(float* __restrict__ p, int n) {
    int i = blockIdx.x * blockDim.x + threadIdx.x;
    if (i < n) p[i] = rsqrtf(p[i]);   // deg >= 1 always (self loop)
}

// Pack W[K,N] f32 -> per-lane WMMA B fragments (f16).
// Fragment for (kt,nt,lane): 16 halfs = B[kt*32 + (lane>>4)*16 + j, nt*16 + (lane&15)]
__global__ void k_pack_w(const float* __restrict__ W, _Float16* __restrict__ Wp,
                         int K, int N) {
    int t = blockIdx.x * blockDim.x + threadIdx.x;
    int ntiles = N >> 4;
    int total = (K >> 5) * ntiles * 32;
    if (t >= total) return;
    int lane = t & 31;
    int tile = t >> 5;
    int nt = tile % ntiles;
    int kt = tile / ntiles;
    int n  = nt * 16 + (lane & 15);
    int kb = kt * 32 + (lane >> 4) * 16;
    _Float16* dstp = Wp + (size_t)t * 16;
    #pragma unroll
    for (int j = 0; j < 16; ++j)
        dstp[j] = (_Float16)W[(size_t)(kb + j) * N + n];
}

// ---------------- WMMA GEMM: C[M,N] = A16[M,K] @ Bpacked, optional bias+relu ----------------
// grid.x = M/16, blockDim.x = (N/16)*32, dynamic LDS = 16*K*2 bytes.
__global__ void k_gemm_wmma(const _Float16* __restrict__ A,
                            const _Float16* __restrict__ Bp,
                            float* __restrict__ C,
                            const float* __restrict__ bias,
                            int M, int N, int K, int relu) {
    extern __shared__ _Float16 sA[];                 // 16 rows x K halfs
    const int tid  = threadIdx.x;
    const int lane = tid & 31;
    const int nt   = tid >> 5;                       // wave id == N tile
    const int m0   = blockIdx.x * 16;

    // cooperative A tile load (16B chunks, coalesced)
    const int kc = K >> 3;                           // 8-half chunks per row
    for (int c = tid; c < 16 * kc; c += blockDim.x) {
        int row = c / kc, c8 = c % kc;
        *(uint4*)(&sA[row * K + c8 * 8]) =
            *(const uint4*)(&A[(size_t)(m0 + row) * K + c8 * 8]);
    }
    __syncthreads();

    const int mlane = lane & 15;
    const int koff  = (lane >> 4) * 16;
    const int ntiles = N >> 4;
    v8f acc = {};
    for (int kt = 0; kt < (K >> 5); ++kt) {
        FragH a, b;
        // A 16x32 f16 fragment (contiguous k = 16*(lane>>4) + s, matching packed B)
        a.u[0] = *(const uint4*)(&sA[mlane * K + kt * 32 + koff]);
        a.u[1] = *(const uint4*)(&sA[mlane * K + kt * 32 + koff + 8]);
        // B fragment: pre-packed contiguous 32B per lane
        const _Float16* bp = Bp + (((size_t)(kt * ntiles + nt) * 32 + lane) * 16);
        b.u[0] = *(const uint4*)(bp);
        b.u[1] = *(const uint4*)(bp + 8);
        acc = __builtin_amdgcn_wmma_f32_16x16x32_f16(false, a.v, false, b.v,
                                                     (short)0, acc, false, false);
    }

    const int colg  = nt * 16 + mlane;
    const int rbase = m0 + (lane >> 4) * 8;          // lanes 16-31 hold M=8..15
    const float bv  = bias ? bias[colg] : 0.0f;
    #pragma unroll
    for (int r = 0; r < 8; ++r) {
        float v = acc[r] + bv;
        if (relu) v = fmaxf(v, 0.0f);
        C[(size_t)(rbase + r) * N + colg] = v;
    }
}

// ---------------- edge scatter: agg[dst] += dinv[s]*dinv[d] * h[src] ----------------
// one wave per edge, 4 floats per lane (D=128)
__global__ void k_scatter(const int* __restrict__ src, const int* __restrict__ dst,
                          const float* __restrict__ dinv, const float* __restrict__ h,
                          float* __restrict__ agg, int E) {
    int e = blockIdx.x * (blockDim.x >> 5) + (threadIdx.x >> 5);
    if (e >= E) return;
    int lane = threadIdx.x & 31;
    int s = src[e], d = dst[e];
    float w = dinv[s] * dinv[d];
    const float4 hv = *(const float4*)&h[(size_t)s * DHID + lane * 4];
    float* ap = &agg[(size_t)d * DHID + lane * 4];
    atomicAdd(ap + 0, w * hv.x);
    atomicAdd(ap + 1, w * hv.y);
    atomicAdd(ap + 2, w * hv.z);
    atomicAdd(ap + 3, w * hv.w);
}

// ---------------- fused self-loop + bias + relu -> f16 next-layer input ----------------
__global__ void k_finish(const float* __restrict__ h, const float* __restrict__ agg,
                         const float* __restrict__ dinv, const float* __restrict__ bias,
                         _Float16* __restrict__ out16, int n_nodes) {
    int i = blockIdx.x * (blockDim.x >> 5) + (threadIdx.x >> 5);
    if (i >= n_nodes) return;
    int lane = threadIdx.x & 31;
    float di = dinv[i];
    float w = di * di;                               // self-loop norm
    size_t base = (size_t)i * DHID + lane * 4;
    float4 hv = *(const float4*)&h[base];
    float4 av = *(const float4*)&agg[base];
    int c = lane * 4;
    v4h o;
    o.x = (_Float16)fmaxf(av.x + w * hv.x + bias[c + 0], 0.0f);
    o.y = (_Float16)fmaxf(av.y + w * hv.y + bias[c + 1], 0.0f);
    o.z = (_Float16)fmaxf(av.z + w * hv.z + bias[c + 2], 0.0f);
    o.w = (_Float16)fmaxf(av.w + w * hv.w + bias[c + 3], 0.0f);
    *(v4h*)&out16[base] = o;
}

// ---------------- pooling ----------------
__global__ void k_pool_sum(const _Float16* __restrict__ h16, const int* __restrict__ batch,
                           float* __restrict__ sums, int n_nodes) {
    int i = blockIdx.x * (blockDim.x >> 5) + (threadIdx.x >> 5);
    if (i >= n_nodes) return;
    int lane = threadIdx.x & 31;
    int g = batch[i];
    v4h hv = *(const v4h*)&h16[(size_t)i * DHID + lane * 4];
    float* sp = &sums[(size_t)g * DHID + lane * 4];
    atomicAdd(sp + 0, (float)hv.x);
    atomicAdd(sp + 1, (float)hv.y);
    atomicAdd(sp + 2, (float)hv.z);
    atomicAdd(sp + 3, (float)hv.w);
}

__global__ void k_counts(const int* __restrict__ batch, float* __restrict__ cnt, int n) {
    int i = blockIdx.x * blockDim.x + threadIdx.x;
    if (i < n) atomicAdd(&cnt[batch[i]], 1.0f);
}

__global__ void k_pool_fin(const float* __restrict__ sums, const float* __restrict__ cnt,
                           _Float16* __restrict__ p16) {
    int i = blockIdx.x * blockDim.x + threadIdx.x;      // NGRAPH*DHID threads
    if (i >= NGRAPH * DHID) return;
    int g = i >> 7;
    float c = fmaxf(cnt[g], 1.0f);
    p16[i] = (_Float16)(sums[i] / c);
}

// ---------------- tiny FC2 (N=26 not WMMA-tileable) ----------------
__global__ void k_fc2(const float* __restrict__ ff, const float* __restrict__ W,
                      const float* __restrict__ b, float* __restrict__ out) {
    int t = blockIdx.x * blockDim.x + threadIdx.x;
    if (t >= NGRAPH * NCLS) return;
    int g = t / NCLS, c = t % NCLS;
    float acc = b[c];
    const float* fg = ff + (size_t)g * DFF;
    for (int k = 0; k < DFF; ++k) acc = fmaf(fg[k], W[(size_t)k * NCLS + c], acc);
    out[t] = acc;
}

// ---------------- host ----------------

extern "C" void kernel_launch(void* const* d_in, const int* in_sizes, int n_in,
                              void* d_out, int out_size, void* d_ws, size_t ws_size,
                              hipStream_t stream) {
    const float* x    = (const float*)d_in[0];
    const int*   ei   = (const int*)d_in[1];
    const int*   bat  = (const int*)d_in[2];
    const float* W1   = (const float*)d_in[3];
    const float* b1   = (const float*)d_in[4];
    const float* W2   = (const float*)d_in[5];
    const float* b2   = (const float*)d_in[6];
    const float* W3   = (const float*)d_in[7];
    const float* b3   = (const float*)d_in[8];
    const float* Wfc  = (const float*)d_in[9];
    const float* bfc  = (const float*)d_in[10];
    const float* Wfc2 = (const float*)d_in[11];
    const float* bfc2 = (const float*)d_in[12];
    float* out = (float*)d_out;

    const int E = in_sizes[1] / 2;
    const int* src = ei;
    const int* dst = ei + E;

    // ---- workspace carve (256B aligned) ----
    char* p = (char*)d_ws;
    auto carve = [&](size_t bytes) -> void* {
        void* r = (void*)p;
        p += (bytes + 255) & ~(size_t)255;
        return r;
    };
    _Float16* x16a = (_Float16*)carve((size_t)NN * DHID * 2);
    _Float16* x16b = (_Float16*)carve((size_t)NN * DHID * 2);
    float*    h    = (float*)   carve((size_t)NN * DHID * 4);
    float*    agg  = (float*)   carve((size_t)NN * DHID * 4);
    float*    dinv = (float*)   carve((size_t)NN * 4);
    _Float16* w1p  = (_Float16*)carve((size_t)DHID * DHID * 2);
    _Float16* w2p  = (_Float16*)carve((size_t)DHID * DHID * 2);
    _Float16* w3p  = (_Float16*)carve((size_t)DHID * DHID * 2);
    _Float16* wfcp = (_Float16*)carve((size_t)DHID * DFF * 2);
    float*    sums = (float*)   carve((size_t)NGRAPH * DHID * 4);
    float*    cnt  = (float*)   carve((size_t)NGRAPH * 4);
    _Float16* p16  = (_Float16*)carve((size_t)NGRAPH * DHID * 2);
    float*    ff1  = (float*)   carve((size_t)NGRAPH * DFF * 4);

    const int TB = 256;
    // convert x -> f16
    k_cvt_f16<<<(NN * DHID + TB - 1) / TB, TB, 0, stream>>>(x, x16a, NN * DHID);
    // pack weights into WMMA B-fragment order
    {
        int t1 = (DHID >> 5) * (DHID >> 4) * 32;        // 1024
        k_pack_w<<<(t1 + TB - 1) / TB, TB, 0, stream>>>(W1, w1p, DHID, DHID);
        k_pack_w<<<(t1 + TB - 1) / TB, TB, 0, stream>>>(W2, w2p, DHID, DHID);
        k_pack_w<<<(t1 + TB - 1) / TB, TB, 0, stream>>>(W3, w3p, DHID, DHID);
        int t2 = (DHID >> 5) * (DFF >> 4) * 32;         // 2048
        k_pack_w<<<(t2 + TB - 1) / TB, TB, 0, stream>>>(Wfc, wfcp, DHID, DFF);
    }
    // symmetric normalization: deg = 1 (self loop) + scatter; dinv = rsqrt(deg)
    k_fill<<<(NN + TB - 1) / TB, TB, 0, stream>>>(dinv, 1.0f, NN);
    k_deg<<<(E + TB - 1) / TB, TB, 0, stream>>>(dst, dinv, E);
    k_rsqrt_inplace<<<(NN + TB - 1) / TB, TB, 0, stream>>>(dinv, NN);

    const _Float16* wp[3] = { w1p, w2p, w3p };
    const float*    bb[3] = { b1, b2, b3 };
    _Float16* bufs[2] = { x16a, x16b };

    const int mtiles    = NN / 16;                       // 6250, exact
    const int waveBlk   = (E + 7) / 8;                   // scatter: 8 waves/block
    const int nodeBlk   = (NN + 7) / 8;

    for (int l = 0; l < 3; ++l) {
        _Float16* in16  = bufs[l & 1];
        _Float16* out16 = bufs[(l + 1) & 1];
        // h = in16 @ Wl   (f16 WMMA, f32 accumulate)
        k_gemm_wmma<<<mtiles, (DHID / 16) * 32, 16 * DHID * 2, stream>>>(
            in16, wp[l], h, nullptr, NN, DHID, DHID, 0);
        hipMemsetAsync(agg, 0, (size_t)NN * DHID * 4, stream);
        k_scatter<<<waveBlk, TB, 0, stream>>>(src, dst, dinv, h, agg, E);
        k_finish<<<nodeBlk, TB, 0, stream>>>(h, agg, dinv, bb[l], out16, NN);
    }

    // global mean pool (batch is graph id per node)
    hipMemsetAsync(sums, 0, (size_t)NGRAPH * DHID * 4, stream);
    hipMemsetAsync(cnt,  0, (size_t)NGRAPH * 4, stream);
    k_pool_sum<<<nodeBlk, TB, 0, stream>>>(bufs[1], bat, sums, NN);   // layer-3 output in x16b
    k_counts<<<(NN + TB - 1) / TB, TB, 0, stream>>>(bat, cnt, NN);
    k_pool_fin<<<(NGRAPH * DHID + TB - 1) / TB, TB, 0, stream>>>(sums, cnt, p16);

    // FC1: [512,128] @ [128,256] + bias, relu  (WMMA)
    k_gemm_wmma<<<NGRAPH / 16, (DFF / 16) * 32, 16 * DHID * 2, stream>>>(
        p16, wfcp, ff1, bfc, NGRAPH, DFF, DHID, 1);
    // FC2: [512,256] @ [256,26] + bias
    k_fc2<<<(NGRAPH * NCLS + TB - 1) / TB, TB, 0, stream>>>(ff1, Wfc2, bfc2, out);
}
